// Model_34651796144565
// MI455X (gfx1250) — matrix operand order repaired and checked by reading, not compiled, not run on hardware
//
#include <hip/hip_runtime.h>
#include <hip/hip_bf16.h>

// ---------------- problem constants (match reference) ----------------
#define N_REQ   100000
#define N_CODE  100000
#define C_IN    256
#define C_OUT   256
#define HEADS   8
#define DIM     32
#define NE      500000
#define E_LAB   200000
#define NEG_SLOPE 0.2f
#define BN_EPS  1e-5f

typedef _Float16 v16h __attribute__((ext_vector_type(16)));
typedef float    v8f  __attribute__((ext_vector_type(8)));

// pack two floats into two f16 halves in one dword (LDS staging format)
__device__ __forceinline__ unsigned pack2h(float a, float b) {
    union { _Float16 h[2]; unsigned u; } x;
    x.h[0] = (_Float16)a; x.h[1] = (_Float16)b;
    return x.u;
}

// order-preserving float <-> uint key for atomic max on floats
__device__ __forceinline__ unsigned fkey(float f) {
    unsigned u = __float_as_uint(f);
    return (u & 0x80000000u) ? ~u : (u | 0x80000000u);
}
__device__ __forceinline__ float funkey(unsigned k) {
    unsigned u = (k & 0x80000000u) ? (k & 0x7FFFFFFFu) : ~k;
    return __uint_as_float(u);
}
#define NEG_INF_KEY 0x007FFFFFu  // fkey(-inf)

// =====================================================================
// 1) Projection GEMM: H[N,256] = X[N,256] @ W^T[256,256] + b
//    WMMA f32_16x16x32_f16, block = 8 waves computes 64 rows x 256 cols.
//    B-fragment loads are double-buffered so DS latency overlaps WMMA.
// =====================================================================
__global__ __launch_bounds__(256) void proj_gemm(
    const float* __restrict__ X, const float* __restrict__ W,
    const float* __restrict__ bias, float* __restrict__ H, int Nrows)
{
    __shared__ unsigned As[64 * 17];   // [m][kpair], pad 17 for bank conflicts
    __shared__ unsigned Bs[256 * 17];  // [n][kpair] where B[k][n] = W[n][k]

    const int tid  = threadIdx.x;
    const int wave = tid >> 5, lane = tid & 31;
    const int lm = lane & 15, hi = lane >> 4;
    const int row0 = blockIdx.x * 64;
    const int mt = wave & 3;            // this wave's 16-row tile
    const int ntBase = (wave >> 2) * 8; // this wave owns 8 of the 16 col-tiles

    v8f acc[8] = {};

    for (int k0 = 0; k0 < C_IN; k0 += 32) {
        // stage A: 64 rows x 16 dword-pairs (= 32 f16 K values)
        #pragma unroll
        for (int i = 0; i < 4; i++) {
            int idx = tid + i * 256;
            int m = idx >> 4, p = idx & 15;
            int r = row0 + m;
            float2 v = (r < Nrows) ? *(const float2*)(X + (size_t)r * C_IN + k0 + 2 * p)
                                   : make_float2(0.f, 0.f);
            As[m * 17 + p] = pack2h(v.x, v.y);
        }
        // stage B: all 256 output cols x 16 pairs  (W row-major: B[k][n] = W[n][k])
        #pragma unroll
        for (int i = 0; i < 16; i++) {
            int idx = tid + i * 256;
            int n = idx >> 4, p = idx & 15;
            float2 v = *(const float2*)(W + (size_t)n * C_IN + k0 + 2 * p);
            Bs[n * 17 + p] = pack2h(v.x, v.y);
        }
        __syncthreads();

        // A fragment (contiguous K order, matches B):
        // lanes0-15: VGPR0-7 -> K0..15 ; lanes16-31: K16..31
        union F { unsigned u[8]; v16h v; };
        F afr;
        const int am = mt * 16 + lm;
        #pragma unroll
        for (int v = 0; v < 8; v++) {
            int p = v + hi * 8;
            afr.u[v] = As[am * 17 + p];
        }

        // B fragment: lane = N, VGPR v half h -> K = 2v+h+16*hi
        // double-buffered: fetch tile t+1 while wmma consumes tile t
        F bfr[2];
        {
            const int bn = ntBase * 16 + lm;
            #pragma unroll
            for (int v = 0; v < 8; v++) bfr[0].u[v] = Bs[bn * 17 + v + hi * 8];
        }
        #pragma unroll
        for (int t = 0; t < 8; t++) {
            const int cur = t & 1, nxt = cur ^ 1;
            if (t < 7) {
                const int bn = (ntBase + t + 1) * 16 + lm;
                #pragma unroll
                for (int v = 0; v < 8; v++) bfr[nxt].u[v] = Bs[bn * 17 + v + hi * 8];
            }
            acc[t] = __builtin_amdgcn_wmma_f32_16x16x32_f16(
                false, afr.v, false, bfr[cur].v, (short)0, acc[t], false, false);
        }
        __syncthreads();
    }

    // C layout: VGPR r -> M = r + 8*hi ; N = lane%16
    #pragma unroll
    for (int t = 0; t < 8; t++) {
        int col = (ntBase + t) * 16 + lm;
        float bc = bias[col];
        #pragma unroll
        for (int r = 0; r < 8; r++) {
            int row = row0 + mt * 16 + r + 8 * hi;
            if (row < Nrows) H[(size_t)row * C_OUT + col] = acc[t][r] + bc;
        }
    }
}

// =====================================================================
// 2) Per-node attention scalars: a[n,h] = sum_d H[n,h*32+d]*att[h,d]
// =====================================================================
__global__ void att_pre(const float* __restrict__ H,
                        const float* __restrict__ attA, const float* __restrict__ attB,
                        float* __restrict__ aA, float* __restrict__ aB, int N)
{
    int idx = blockIdx.x * blockDim.x + threadIdx.x;
    if (idx >= N * HEADS) return;
    int n = idx >> 3, h = idx & 7;
    const float* hp = H + (size_t)n * C_OUT + h * DIM;
    const float* a1 = attA + h * DIM;
    const float* a2 = attB + h * DIM;
    float s1 = 0.f, s2 = 0.f;
    #pragma unroll
    for (int d = 0; d < DIM; d++) { float v = hp[d]; s1 += v * a1[d]; s2 += v * a2[d]; }
    aA[idx] = s1; aB[idx] = s2;
}

// =====================================================================
// 3) Segment softmax pass A: logits + atomic segment-max (uint-key trick)
// =====================================================================
__global__ void att_logit_max(const float* __restrict__ aS, const float* __restrict__ aD,
                              const int* __restrict__ ei, int E,
                              unsigned* __restrict__ mkey, float* __restrict__ logit)
{
    int idx = blockIdx.x * blockDim.x + threadIdx.x;
    if (idx >= E * HEADS) return;
    int e = idx >> 3, h = idx & 7;
    int src = ei[e], dst = ei[E + e];
    float l = aS[src * HEADS + h] + aD[dst * HEADS + h];
    l = (l > 0.f) ? l : NEG_SLOPE * l;          // leaky_relu
    logit[idx] = l;
    atomicMax(&mkey[dst * HEADS + h], fkey(l));
}

// pass B: exp(logit - max) in place + atomic segment-sum
__global__ void att_exp_sum(float* __restrict__ logit_eexp, const int* __restrict__ ei,
                            int E, const unsigned* __restrict__ mkey,
                            float* __restrict__ ssum)
{
    int idx = blockIdx.x * blockDim.x + threadIdx.x;
    if (idx >= E * HEADS) return;
    int e = idx >> 3, h = idx & 7;
    int dst = ei[E + e];
    float m  = funkey(mkey[dst * HEADS + h]);   // finite for every dst with an edge
    float ex = expf(logit_eexp[idx] - m);
    logit_eexp[idx] = ex;
    atomicAdd(&ssum[dst * HEADS + h], ex);
}

// pass C: out[dst] += alpha[e,h] * Hsrc[src]  (one wave32 per edge; lane = dim%32)
__global__ __launch_bounds__(256) void msg_agg(
    const float* __restrict__ Hsrc, const int* __restrict__ ei, int E,
    const float* __restrict__ eexp, const float* __restrict__ ssum,
    float* __restrict__ out)
{
    int lane = threadIdx.x & 31;
    int e = blockIdx.x * 8 + (threadIdx.x >> 5);
    if (e >= E) return;
    int src = ei[e], dst = ei[E + e];
    const float* hs = Hsrc + (size_t)src * C_OUT;
    float* op = out + (size_t)dst * C_OUT;
    // early prefetch of the gathered source row (global_prefetch_b8)
    __builtin_prefetch(hs + lane, 0, 0);
    __builtin_prefetch(hs + 128 + lane, 0, 0);
    #pragma unroll
    for (int k = 0; k < HEADS; k++) {
        float w = eexp[(size_t)e * HEADS + k] / (ssum[dst * HEADS + k] + 1e-16f);
        atomicAdd(op + k * DIM + lane, hs[k * DIM + lane] * w);
    }
}

// =====================================================================
// 4) BatchNorm stats over relu(out), coalesced two-phase reduction.
//    Phase 1: thread t owns column t; block owns a 256-row slab; lane
//    accesses are unit-stride (coalesced). Partial sums -> atomicAdd.
//    (semantic attention _group is identity for a single metapath:
//     softmax over a length-1 vector == 1.0, so z == relu-aggregated out)
// =====================================================================
__global__ __launch_bounds__(256) void bn_partial(const float* __restrict__ Z, int N,
                                                  float* __restrict__ acc /*[512]*/)
{
    int c = threadIdx.x;
    int r0 = blockIdx.x * 256;
    int rend = min(r0 + 256, N);
    float s = 0.f, ss = 0.f;
    #pragma unroll 4
    for (int r = r0; r < rend; ++r) {
        float v = fmaxf(Z[(size_t)r * C_OUT + c], 0.f);  // relu fused here
        s += v; ss += v * v;
    }
    atomicAdd(&acc[c], s);
    atomicAdd(&acc[C_OUT + c], ss);
}

__global__ void bn_finalize(const float* __restrict__ acc, int N,
                            float* __restrict__ mu, float* __restrict__ rstd)
{
    int c = threadIdx.x;  // one block of 256
    float m = acc[c] / (float)N;
    float var = acc[C_OUT + c] / (float)N - m * m;   // biased var, matches jnp.var
    mu[c] = m;
    rstd[c] = rsqrtf(var + BN_EPS);
}

// =====================================================================
// 5) Edge classifier: sigmoid(dot(bn(relu(zr[i])), bn(relu(zc[j]))))
//    one wave32 per labeled edge, shuffle reduction
// =====================================================================
__global__ __launch_bounds__(256) void edge_cls(
    const float* __restrict__ Zr, const float* __restrict__ Zc,
    const int* __restrict__ eli, int EL,
    const float* __restrict__ mur, const float* __restrict__ rsr,
    const float* __restrict__ muc, const float* __restrict__ rsc,
    const float* __restrict__ gamma, const float* __restrict__ beta,
    float* __restrict__ out)
{
    int lane = threadIdx.x & 31;
    int e = blockIdx.x * 8 + (threadIdx.x >> 5);
    if (e >= EL) return;
    int i = eli[e], j = eli[EL + e];
    const float* zr = Zr + (size_t)i * C_OUT;
    const float* zc = Zc + (size_t)j * C_OUT;
    // early prefetch of both gathered rows
    __builtin_prefetch(zr + lane, 0, 0);
    __builtin_prefetch(zc + lane, 0, 0);
    __builtin_prefetch(zr + 128 + lane, 0, 0);
    __builtin_prefetch(zc + 128 + lane, 0, 0);
    float accv = 0.f;
    #pragma unroll
    for (int k = 0; k < 8; k++) {
        int c = k * 32 + lane;
        float vr = (fmaxf(zr[c], 0.f) - mur[c]) * rsr[c] * gamma[c] + beta[c];
        float vc = (fmaxf(zc[c], 0.f) - muc[c]) * rsc[c] * gamma[c] + beta[c];
        accv += vr * vc;
    }
    #pragma unroll
    for (int off = 16; off; off >>= 1) accv += __shfl_down(accv, off, 32);
    if (lane == 0) out[e] = 1.f / (1.f + expf(-accv));
}

// generic fill (also used to zero float buffers: 0x00000000 == 0.0f)
__global__ void fill_u32(unsigned* __restrict__ p, unsigned val, long long n)
{
    long long i = (long long)blockIdx.x * blockDim.x + threadIdx.x;
    long long stride = (long long)gridDim.x * blockDim.x;
    for (; i < n; i += stride) p[i] = val;
}

// =====================================================================
extern "C" void kernel_launch(void* const* d_in, const int* in_sizes, int n_in,
                              void* d_out, int out_size, void* d_ws, size_t ws_size,
                              hipStream_t stream)
{
    const float* x_req  = (const float*)d_in[0];
    const float* x_code = (const float*)d_in[1];
    const int*   ei_rc  = (const int*)d_in[2];   // [2,E] src=req row0, dst=code row1
    const int*   ei_cr  = (const int*)d_in[3];   // [2,E] src=code row0, dst=req row1
    const int*   eli    = (const int*)d_in[4];   // [2,E_LAB]
    const float* W_req  = (const float*)d_in[5];
    const float* b_req  = (const float*)d_in[6];
    const float* W_code = (const float*)d_in[7];
    const float* b_code = (const float*)d_in[8];
    const float* att_src_rc = (const float*)d_in[9];
    const float* att_dst_rc = (const float*)d_in[10];
    const float* att_src_cr = (const float*)d_in[11];
    const float* att_dst_cr = (const float*)d_in[12];
    // d_in[13..15] (k_W, k_b, q): semantic attention over ONE metapath is
    // identity (softmax of a singleton == 1.0) -> unused.
    const float* gamma = (const float*)d_in[16];
    const float* beta  = (const float*)d_in[17];
    float* out = (float*)d_out;

    // ---- workspace carve-up ----
    char* ws = (char*)d_ws;
    size_t off = 0;
    auto carve = [&](size_t bytes) -> char* {
        char* p = ws + off;
        off += (bytes + 255) & ~(size_t)255;
        return p;
    };
    float*    h_req    = (float*)carve((size_t)N_REQ  * C_OUT * 4);
    float*    h_code   = (float*)carve((size_t)N_CODE * C_OUT * 4);
    float*    out_req  = (float*)carve((size_t)N_REQ  * C_OUT * 4);
    float*    out_code = (float*)carve((size_t)N_CODE * C_OUT * 4);
    float*    aS_rc = (float*)carve((size_t)N_REQ  * HEADS * 4);  // req as src (rc)
    float*    aD_cr = (float*)carve((size_t)N_REQ  * HEADS * 4);  // req as dst (cr)
    float*    aD_rc = (float*)carve((size_t)N_CODE * HEADS * 4);  // code as dst (rc)
    float*    aS_cr = (float*)carve((size_t)N_CODE * HEADS * 4);  // code as src (cr)
    unsigned* mkey  = (unsigned*)carve((size_t)N_REQ * HEADS * 4); // reused per edge type
    float*    ssum  = (float*)carve((size_t)N_REQ * HEADS * 4);    // reused per edge type
    float*    eexp  = (float*)carve((size_t)NE * HEADS * 4);       // logits -> exp, in place
    float*    bnacc = (float*)carve(2 * C_OUT * 4);                // reused per node type
    float*    mu_r  = (float*)carve(C_OUT * 4);
    float*    rs_r  = (float*)carve(C_OUT * 4);
    float*    mu_c  = (float*)carve(C_OUT * 4);
    float*    rs_c  = (float*)carve(C_OUT * 4);

    const int TB = 256;

    // 1) projections (WMMA)
    int gGemm = (N_REQ + 63) / 64;
    proj_gemm<<<gGemm, TB, 0, stream>>>(x_req,  W_req,  b_req,  h_req,  N_REQ);
    proj_gemm<<<gGemm, TB, 0, stream>>>(x_code, W_code, b_code, h_code, N_CODE);

    // 2) per-node attention scalars
    int gAtt = (N_REQ * HEADS + TB - 1) / TB;
    att_pre<<<gAtt, TB, 0, stream>>>(h_req,  att_src_rc, att_dst_cr, aS_rc, aD_cr, N_REQ);
    att_pre<<<gAtt, TB, 0, stream>>>(h_code, att_dst_rc, att_src_cr, aD_rc, aS_cr, N_CODE);

    const int gEH = (NE * HEADS + TB - 1) / TB;   // per-(edge,head)
    const int gE  = (NE + 7) / 8;                 // wave-per-edge

    // 3a) edge type rc: req -> code (dst = code)
    fill_u32<<<2048, TB, 0, stream>>>(mkey, NEG_INF_KEY, (long long)N_CODE * HEADS);
    fill_u32<<<2048, TB, 0, stream>>>((unsigned*)ssum, 0u, (long long)N_CODE * HEADS);
    fill_u32<<<4096, TB, 0, stream>>>((unsigned*)out_code, 0u, (long long)N_CODE * C_OUT);
    att_logit_max<<<gEH, TB, 0, stream>>>(aS_rc, aD_rc, ei_rc, NE, mkey, eexp);
    att_exp_sum  <<<gEH, TB, 0, stream>>>(eexp, ei_rc, NE, mkey, ssum);
    msg_agg      <<<gE,  TB, 0, stream>>>(h_req, ei_rc, NE, eexp, ssum, out_code);

    // 3b) edge type cr: code -> req (dst = req)
    fill_u32<<<2048, TB, 0, stream>>>(mkey, NEG_INF_KEY, (long long)N_REQ * HEADS);
    fill_u32<<<2048, TB, 0, stream>>>((unsigned*)ssum, 0u, (long long)N_REQ * HEADS);
    fill_u32<<<4096, TB, 0, stream>>>((unsigned*)out_req, 0u, (long long)N_REQ * C_OUT);
    att_logit_max<<<gEH, TB, 0, stream>>>(aS_cr, aD_cr, ei_cr, NE, mkey, eexp);
    att_exp_sum  <<<gEH, TB, 0, stream>>>(eexp, ei_cr, NE, mkey, ssum);
    msg_agg      <<<gE,  TB, 0, stream>>>(h_code, ei_cr, NE, eexp, ssum, out_req);

    // 4) batch-norm stats (relu fused), coalesced two-phase
    int gBn = (N_REQ + 255) / 256;
    fill_u32<<<1, 512, 0, stream>>>((unsigned*)bnacc, 0u, 2 * C_OUT);
    bn_partial<<<gBn, TB, 0, stream>>>(out_req, N_REQ, bnacc);
    bn_finalize<<<1, C_OUT, 0, stream>>>(bnacc, N_REQ, mu_r, rs_r);
    fill_u32<<<1, 512, 0, stream>>>((unsigned*)bnacc, 0u, 2 * C_OUT);
    bn_partial<<<gBn, TB, 0, stream>>>(out_code, N_CODE, bnacc);
    bn_finalize<<<1, C_OUT, 0, stream>>>(bnacc, N_CODE, mu_c, rs_c);

    // 5) classifier
    int gCls = (E_LAB + 7) / 8;
    edge_cls<<<gCls, TB, 0, stream>>>(out_req, out_code, eli, E_LAB,
                                      mu_r, rs_r, mu_c, rs_c, gamma, beta, out);
    (void)in_sizes; (void)n_in; (void)out_size; (void)ws_size;
}
